// FFTLayer_80917183857181
// MI455X (gfx1250) — matrix operand order repaired and checked by reading, not compiled, not run on hardware
//
#include <hip/hip_runtime.h>
#include <hip/hip_bf16.h>

typedef __attribute__((ext_vector_type(16))) _Float16 v16h;
typedef __attribute__((ext_vector_type(2)))  _Float16 v2h;
typedef __attribute__((ext_vector_type(8)))  float    v8f;

#define FRAME_LEN   512
#define ROWS_TOTAL  (4096 * 32)          // 131072 rows of 512-point DFT
#define COLTILES    17                   // freqs 0..271 computed; 0..256 stored (+mirror)
#define KSTEPS      16                   // 512 / 32
#define THREADS     256                  // 8 waves of 32
#define ROWS_PER_WG 256
#define NUM_WG      (ROWS_TOTAL / ROWS_PER_WG)   // 512

// Swizzled B in LDS, ks-major so the whole per-kstep chain uses immediate DS offsets:
// chunkIdx = ((ks*COLTILES + t)*2 + c)*32 + L   -> one uint4 (16B) per chunk
//   t stride = 128 uint4 (2048 B), c stride = 32 uint4 (512 B), max imm = 33280 < 64K
#define BBYTES   (COLTILES * KSTEPS * 2 * 32 * 16)   // 278528
#define SMEM_SZ  (BBYTES + 2048 + 2048)              // + cos LUT + window

extern "C" __global__ __launch_bounds__(THREADS, 1)
void fft512_real_wmma(const float* __restrict__ x,
                      const float* __restrict__ win,
                      float* __restrict__ out) {
  extern __shared__ unsigned char smem[];
  uint4*    Bv   = (uint4*)smem;
  unsigned* Bu   = (unsigned*)smem;
  float*    ctab = (float*)(smem + BBYTES);          // 512 f32 cos LUT
  float*    wl   = (float*)(smem + BBYTES + 2048);   // 512 f32 window

  const int tid = threadIdx.x;

  // --- Stage 0: cos LUT + window into LDS ---
  for (int i = tid; i < 512; i += THREADS) {
    ctab[i] = __builtin_cosf((float)i * (6.28318530717958647692f / 512.0f));
    wl[i]   = win[i];
  }
  __syncthreads();

  // --- Stage 1: build swizzled f16 B fragments (ks-major layout) ---
  // B[s,k] = cos(2*pi*s*k/512) = ctab[(s*k) & 511]
  // B-fragment per lane: N = L&15, lane half hB = L>>4 selects K 8hB..8hB+7 / 16+8hB..16+8hB+7
  // within a 32-K step; slot p (0..15) -> K = 8*hB + p (p<8), 16 + 8*hB + p-8 (p>=8); chunk c holds p = c*8 + j.
  const int NU32 = BBYTES / 4;           // 69632 packed f16 pairs
  for (int e = tid; e < NU32; e += THREADS) {
    int jp   = e & 3;                    // pair within 16B chunk (j = 2*jp, 2*jp+1)
    int L    = (e >> 2) & 31;
    int c    = (e >> 7) & 1;
    int idx2 = e >> 8;                   // ks*COLTILES + t, 0..271
    int ks   = idx2 / COLTILES;
    int t    = idx2 - ks * COLTILES;
    int hB   = L >> 4;
    int s    = ks * 32 + hB * 8 + c * 16 + jp * 2;   // sample index (K dim)
    int f    = t * 16 + (L & 15);                    // frequency column (N dim)
    float c0 = ctab[(s * f) & 511];
    float c1 = ctab[((s + 1) * f) & 511];
    v2h p; p[0] = (_Float16)c0; p[1] = (_Float16)c1;
    Bu[e] = __builtin_bit_cast(unsigned, p);
  }
  __syncthreads();

  // --- Stage 2: WMMA GEMM, 17 accumulator tiles per wave, pipelined loads ---
  const int lane = tid & 31;
  const int wv   = tid >> 5;             // wave id 0..7
  const int h    = lane >> 4;            // A-fragment lane half
  const int n    = lane & 15;            // M (A) / N (C) position
  const long wgRow = (long)blockIdx.x * ROWS_PER_WG;
  const float4* wl4 = (const float4*)wl;

  #pragma unroll 1
  for (int rt = 0; rt < 2; ++rt) {
    const long rBase = wgRow + (long)wv * 32 + (long)rt * 16;
    const float4* rowp = (const float4*)(x + (rBase + n) * FRAME_LEN);

    // Pull the next row tile toward the WGP while this one is consumed.
    if (rt == 0) {
      const float* nextp = x + (rBase + 16 + n) * FRAME_LEN;
      __builtin_prefetch(nextp, 0, 1);
      __builtin_prefetch(nextp + 256, 0, 1);
    }

    v8f C[COLTILES] = {};                // 17 * 8 = 136 f32 accumulators

    // A-fragment float4 indices for this lane half (ISA 16-bit A 16x32 layout):
    // slots 0..7 <- K = 8h..8h+7, slots 8..15 <- K = 16+8h..16+8h+7
    float4 A0 = rowp[h * 2],  A1 = rowp[h * 2 + 1];
    float4 A2 = rowp[h * 2 + 4], A3 = rowp[h * 2 + 5];

    #pragma unroll 1
    for (int ks = 0; ks < KSTEPS; ++ks) {
      const int i0 = ks * 8 + h * 2;
      float4 W0 = wl4[i0], W1 = wl4[i0 + 1], W2 = wl4[i0 + 4], W3 = wl4[i0 + 5];
      float fa[16] = {A0.x*W0.x, A0.y*W0.y, A0.z*W0.z, A0.w*W0.w,
                      A1.x*W1.x, A1.y*W1.y, A1.z*W1.z, A1.w*W1.w,
                      A2.x*W2.x, A2.y*W2.y, A2.z*W2.z, A2.w*W2.w,
                      A3.x*W3.x, A3.y*W3.y, A3.z*W3.z, A3.w*W3.w};
      v16h a;
      #pragma unroll
      for (int p = 0; p < 16; ++p) a[p] = (_Float16)fa[p];

      // Prefetch next k-step's A (global) behind the 17-WMMA chain.
      if (ks < KSTEPS - 1) {
        const int i1 = i0 + 8;
        A0 = rowp[i1]; A1 = rowp[i1 + 1]; A2 = rowp[i1 + 4]; A3 = rowp[i1 + 5];
      }

      // B chain: ks-major base, all offsets are DS immediates; double-buffered.
      const int ci = ks * (COLTILES * 2 * 32) + lane;   // uint4 index
      uint4 lo = Bv[ci], hi = Bv[ci + 32];
      #pragma unroll
      for (int t = 0; t < COLTILES; ++t) {
        uint4 lo_n, hi_n;
        if (t < COLTILES - 1) {                          // load t+1 before using t
          lo_n = Bv[ci + (t + 1) * 128];
          hi_n = Bv[ci + (t + 1) * 128 + 32];
        }
        struct U2 { uint4 a, b; } u{lo, hi};
        v16h b = __builtin_bit_cast(v16h, u);
        C[t] = __builtin_amdgcn_wmma_f32_16x16x32_f16(
            /*neg_a=*/false, a, /*neg_b=*/false, b,
            /*c_mod=*/(short)0, C[t], /*reuse_a=*/false, /*reuse_b=*/false);
        lo = lo_n; hi = hi_n;
      }
    }

    // Stores: direct k<=256, mirror 512-k for 1<=k<=255 (Re(X) symmetry)
    #pragma unroll
    for (int t = 0; t < COLTILES; ++t) {
      const int k = t * 16 + n;
      #pragma unroll
      for (int v = 0; v < 8; ++v) {
        const long row = rBase + v + 8 * h;       // C layout: VGPR v -> M = v + 8*laneHalf
        const float val = C[t][v];
        if (k <= 256)           out[row * FRAME_LEN + k] = val;
        if (k >= 1 && k <= 255) out[row * FRAME_LEN + (FRAME_LEN - k)] = val;
      }
    }
  }
}

extern "C" void kernel_launch(void* const* d_in, const int* in_sizes, int n_in,
                              void* d_out, int out_size, void* d_ws, size_t ws_size,
                              hipStream_t stream) {
  (void)in_sizes; (void)n_in; (void)out_size; (void)d_ws; (void)ws_size;
  const float* x   = (const float*)d_in[0];
  const float* win = (const float*)d_in[1];
  float* out = (float*)d_out;

  // Opt in to the large (320KB/WGP) CDNA5 LDS allocation.
  static_assert(SMEM_SZ <= 320 * 1024, "fits WGP LDS");
  hipFuncSetAttribute((const void*)fft512_real_wmma,
                      hipFuncAttributeMaxDynamicSharedMemorySize, SMEM_SZ);

  fft512_real_wmma<<<dim3(NUM_WG), dim3(THREADS), SMEM_SZ, stream>>>(x, win, out);
}
